// SingleMambaBlock_46127948759306
// MI455X (gfx1250) — matrix operand order repaired and checked by reading, not compiled, not run on hardware
//
#include <hip/hip_runtime.h>

// ---------------------------------------------------------------------------
// SingleMambaBlock for MI455X (gfx1250), wave32, WMMA bf16 path.
// B=4, L=2048, Dm=1024, E=2048, N=16, R=64, K_conv=4
// ---------------------------------------------------------------------------

#define Bb   4
#define Ll   2048
#define Dm   1024
#define Ee   2048
#define Nn   16
#define Rr   64
#define BL   (Bb * Ll)            // 8192

typedef __bf16 bf16_t;
typedef bf16_t bf16x8  __attribute__((ext_vector_type(8)));
typedef bf16_t bf16x16 __attribute__((ext_vector_type(16)));
typedef float  f32x8   __attribute__((ext_vector_type(8)));

__device__ __forceinline__ unsigned short f32_to_bf16u(float f) {
    unsigned int u = __float_as_uint(f);
    unsigned int r = u + 0x7FFFu + ((u >> 16) & 1u);   // round-to-nearest-even
    return (unsigned short)(r >> 16);
}
__device__ __forceinline__ float bf16u_to_f32(unsigned short u) {
    return __uint_as_float(((unsigned int)u) << 16);
}

// --- WMMA fragment loaders (16-bit, 16x16x32; layouts per CDNA5 ISA) --
// A (16x32, row m = lane%16): lanes 0-15 hold K={0..7,16..23}, lanes 16-31 K={8..15,24..31}
__device__ __forceinline__ bf16x16 load_frag_a(const bf16_t* __restrict__ row, int k0, int hlf) {
    bf16x8 lo = *(const bf16x8*)(row + k0 + 8 * hlf);
    bf16x8 hi = *(const bf16x8*)(row + k0 + 16 + 8 * hlf);
    return __builtin_shufflevector(lo, hi, 0,1,2,3,4,5,6,7,8,9,10,11,12,13,14,15);
}
// B (32x16, col n = lane%16): lanes 0-15 hold K={0..7,16..23}, lanes 16-31 K={8..15,24..31}
__device__ __forceinline__ bf16x16 load_frag_b(const bf16_t* __restrict__ row, int k0, int hlf) {
    bf16x8 lo = *(const bf16x8*)(row + k0 + 8 * hlf);
    bf16x8 hi = *(const bf16x8*)(row + k0 + 16 + 8 * hlf);
    return __builtin_shufflevector(lo, hi, 0,1,2,3,4,5,6,7,8,9,10,11,12,13,14,15);
}

// ---------------------------------------------------------------------------
// Kernel 1: residual = x + residual ; xn = LayerNorm(residual) -> bf16
// ---------------------------------------------------------------------------
__global__ __launch_bounds__(256) void resln_kernel(
    const float* __restrict__ x, const float* __restrict__ res_in,
    const float* __restrict__ gam, const float* __restrict__ bet,
    float* __restrict__ res_out, unsigned short* __restrict__ xn_bf)
{
    const int row = blockIdx.x;           // b*L + l
    const int tid = threadIdx.x;
    const float* xr = x      + (size_t)row * Dm;
    const float* rr = res_in + (size_t)row * Dm;

    float v[4]; float s = 0.f, ss = 0.f;
#pragma unroll
    for (int i = 0; i < 4; ++i) {
        float t = xr[tid + 256 * i] + rr[tid + 256 * i];
        v[i] = t; s += t; ss += t * t;
    }
    __shared__ float s1[256], s2[256];
    s1[tid] = s; s2[tid] = ss; __syncthreads();
    for (int off = 128; off > 0; off >>= 1) {
        if (tid < off) { s1[tid] += s1[tid + off]; s2[tid] += s2[tid + off]; }
        __syncthreads();
    }
    const float mu   = s1[0] * (1.f / Dm);
    const float var  = s2[0] * (1.f / Dm) - mu * mu;
    const float rstd = rsqrtf(var + 1e-5f);

    float*          ro = res_out + (size_t)row * Dm;
    unsigned short* xo = xn_bf   + (size_t)row * Dm;
#pragma unroll
    for (int i = 0; i < 4; ++i) {
        int c = tid + 256 * i;
        ro[c] = v[i];
        float xn = (v[i] - mu) * rstd * gam[c] + bet[c];
        xo[c] = f32_to_bf16u(xn);
    }
}

// ---------------------------------------------------------------------------
// Kernel 2: fp32 -> bf16 cast (weights)
// ---------------------------------------------------------------------------
__global__ void cast_f32_bf16(const float* __restrict__ src,
                              unsigned short* __restrict__ dst, int n)
{
    int i = blockIdx.x * blockDim.x + threadIdx.x;
    if (i < n) dst[i] = f32_to_bf16u(src[i]);
}

// ---------------------------------------------------------------------------
// Kernel 3: WMMA GEMM  C[M,N] = A[M,K](bf16) * W[N,K]^T(bf16)
// 256 threads = 8 waves arranged 4(M) x 2(N); each wave computes 16 x (JT*16).
// JT is a compile-time tile count -> straight-line K-loop, no EXEC masking,
// so v_wmma runs with EXEC all-ones and no accumulator shuffling.
// Grid: x = N / (2*JT*16), y = M / 64.  (All N here divisible by 2*JT*16.)
// ---------------------------------------------------------------------------
template<int JT, bool STORE_BF16>
__global__ __launch_bounds__(256) void wmma_gemm_bf16(
    const unsigned short* __restrict__ Au, const unsigned short* __restrict__ Wu,
    float* __restrict__ Cf, unsigned short* __restrict__ Cb,
    int M, int N, int K)
{
    const int lane = threadIdx.x & 31;
    const int wid  = threadIdx.x >> 5;
    const int hlf  = lane >> 4;
    const int l15  = lane & 15;
    const int mtile  = blockIdx.y * 64 + (wid & 3) * 16;
    const int ntile0 = (blockIdx.x * 2 + (wid >> 2)) * (JT * 16);

    const bf16_t* A = (const bf16_t*)Au;
    const bf16_t* W = (const bf16_t*)Wu;

    f32x8 acc[JT] = {};
    const bf16_t* arow = A + (size_t)(mtile + l15) * K;
    const bf16_t* wrow[JT];
#pragma unroll
    for (int j = 0; j < JT; ++j)
        wrow[j] = W + (size_t)(ntile0 + j * 16 + l15) * K;

    for (int k0 = 0; k0 < K; k0 += 32) {
        __builtin_prefetch((const void*)(arow + k0 + 256), 0, 3); // global_prefetch_b8 (near)
        // issue all fragment loads first, then the JT matrix ops
        bf16x16 afrag = load_frag_a(arow, k0, hlf);
        bf16x16 bfrag[JT];
#pragma unroll
        for (int j = 0; j < JT; ++j) bfrag[j] = load_frag_b(wrow[j], k0, hlf);
#pragma unroll
        for (int j = 0; j < JT; ++j)
            acc[j] = __builtin_amdgcn_wmma_f32_16x16x32_bf16(
                false, afrag, false, bfrag[j], (short)0, acc[j], false, false);
    }

    // C/D layout: VGPR r -> lanes 0-15: (M=r, N=lane); lanes 16-31: (M=r+8, N=lane-16)
#pragma unroll
    for (int j = 0; j < JT; ++j) {
        const int n = ntile0 + j * 16 + l15;
#pragma unroll
        for (int r = 0; r < 8; ++r) {
            const int m = mtile + hlf * 8 + r;
            const size_t idx = (size_t)m * N + n;
            if (STORE_BF16) Cb[idx] = f32_to_bf16u(acc[j][r]);
            else            Cf[idx] = acc[j][r];
        }
    }
}

// ---------------------------------------------------------------------------
// Kernel 4: causal depthwise conv1d (K=4) + bias + SiLU on xi = xz[:, :E]
// ---------------------------------------------------------------------------
__global__ __launch_bounds__(256) void conv_silu_kernel(
    const unsigned short* __restrict__ xz,   // (BL, 2E) bf16
    const float* __restrict__ cw,            // (E,1,4)
    const float* __restrict__ cb,            // (E)
    unsigned short* __restrict__ xc)         // (BL, E) bf16
{
    size_t i = (size_t)blockIdx.x * blockDim.x + threadIdx.x;  // BL*E elements
    if (i >= (size_t)BL * Ee) return;
    int    e  = (int)(i & (Ee - 1));
    size_t bl = i >> 11;                     // E = 2^11
    int    l  = (int)(bl & (Ll - 1));

    float acc = cb[e];
#pragma unroll
    for (int k = 0; k < 4; ++k) {
        int ls = l - 3 + k;
        if (ls >= 0) {
            float xv = bf16u_to_f32(xz[(bl - (size_t)l + ls) * (2 * Ee) + e]);
            acc = fmaf(cw[e * 4 + k], xv, acc);
        }
    }
    float s = acc / (1.f + __expf(-acc));    // SiLU
    xc[i] = f32_to_bf16u(s);
}

// ---------------------------------------------------------------------------
// Kernel 5: extract dt-rank columns of dbl -> bf16 (GEMM3 input)
// ---------------------------------------------------------------------------
__global__ void extract_dt_kernel(const float* __restrict__ dbl,
                                  unsigned short* __restrict__ dtr)
{
    int i = blockIdx.x * blockDim.x + threadIdx.x;   // BL*R
    if (i >= BL * Rr) return;
    int m = i >> 6, r = i & 63;
    dtr[i] = f32_to_bf16u(dbl[(size_t)m * (Rr + 2 * Nn) + r]);
}

// ---------------------------------------------------------------------------
// Kernel 6: selective scan, channel-parallel over (b,e), sequential over L.
// h[N=16] in registers; B_t/C_t broadcast via LDS; fuses softplus(dt+bias),
// D-skip and SiLU(z) gating. Output y -> bf16 for the final WMMA GEMM.
// ---------------------------------------------------------------------------
__global__ __launch_bounds__(256) void scan_kernel(
    const float* __restrict__ dbl,             // (BL, R+2N)
    const unsigned short* __restrict__ dt_bf,  // (BL, E)  raw dt_proj output
    const unsigned short* __restrict__ xc_bf,  // (BL, E)
    const unsigned short* __restrict__ xz_bf,  // (BL, 2E), z at +E
    const float* __restrict__ dt_bias,         // (E)
    const float* __restrict__ A_log,           // (E, N)
    const float* __restrict__ D_skip,          // (E)
    unsigned short* __restrict__ y_bf)         // (BL, E)
{
    const int b = blockIdx.x;                        // 0..3
    const int e = blockIdx.y * 256 + threadIdx.x;    // 0..E-1

    float a[Nn], h[Nn];
#pragma unroll
    for (int n = 0; n < Nn; ++n) { a[n] = -__expf(A_log[e * Nn + n]); h[n] = 0.f; }
    const float dtb = dt_bias[e];
    const float dsk = D_skip[e];

    __shared__ float sB[Nn], sC[Nn];
    const size_t base = (size_t)b * Ll;

    for (int l = 0; l < Ll; ++l) {
        const size_t bl = base + l;
        __syncthreads();
        if (threadIdx.x < 2 * Nn) {
            float v = dbl[bl * (Rr + 2 * Nn) + Rr + threadIdx.x];
            if (threadIdx.x < Nn) sB[threadIdx.x] = v;
            else                  sC[threadIdx.x - Nn] = v;
        }
        __syncthreads();

        const size_t idx = bl * Ee + e;
        float dtv = bf16u_to_f32(dt_bf[idx]) + dtb;
        dtv = (dtv > 20.f) ? dtv : __logf(1.f + __expf(dtv));   // softplus
        float xv = bf16u_to_f32(xc_bf[idx]);
        float dx = dtv * xv;

        float acc = 0.f;
#pragma unroll
        for (int n = 0; n < Nn; ++n) {
            float dA = __expf(dtv * a[n]);
            h[n] = fmaf(dA, h[n], dx * sB[n]);
            acc  = fmaf(h[n], sC[n], acc);
        }
        float y = fmaf(xv, dsk, acc);
        float z = bf16u_to_f32(xz_bf[bl * (2 * Ee) + Ee + e]);
        y *= z / (1.f + __expf(-z));                            // * SiLU(z)
        y_bf[idx] = f32_to_bf16u(y);
    }
}

// ---------------------------------------------------------------------------
// Launch
// ---------------------------------------------------------------------------
extern "C" void kernel_launch(void* const* d_in, const int* in_sizes, int n_in,
                              void* d_out, int out_size, void* d_ws, size_t ws_size,
                              hipStream_t stream) {
    const float* x        = (const float*)d_in[0];
    const float* residual = (const float*)d_in[1];
    const float* ln_w     = (const float*)d_in[2];
    const float* ln_b     = (const float*)d_in[3];
    const float* in_proj  = (const float*)d_in[4];   // (2E, Dm)
    const float* conv_w   = (const float*)d_in[5];   // (E,1,4)
    const float* conv_b   = (const float*)d_in[6];   // (E)
    const float* x_proj   = (const float*)d_in[7];   // (R+2N, E)
    const float* dt_proj  = (const float*)d_in[8];   // (E, R)
    const float* dt_bias  = (const float*)d_in[9];   // (E)
    const float* A_log    = (const float*)d_in[10];  // (E, N)
    const float* D_skip   = (const float*)d_in[11];  // (E)
    const float* out_proj = (const float*)d_in[12];  // (Dm, E)

    float* out_f = (float*)d_out;                     // (BL, Dm)
    float* res_f = (float*)d_out + (size_t)BL * Dm;   // (BL, Dm)

    // ---- workspace layout (bytes) ----
    char* ws = (char*)d_ws;
    unsigned short* xn_bf  = (unsigned short*)(ws + 0);                  // BL*Dm
    unsigned short* w_in   = (unsigned short*)(ws + 16777216);           // 2E*Dm
    unsigned short* w_xp   = (unsigned short*)(ws + 25165824);           // 96*E
    unsigned short* w_dt   = (unsigned short*)(ws + 25559040);           // E*R
    unsigned short* w_out  = (unsigned short*)(ws + 25821184);           // Dm*E
    unsigned short* xz_bf  = (unsigned short*)(ws + 30015488);           // BL*2E
    unsigned short* xc_bf  = (unsigned short*)(ws + 97124352);           // BL*E
    float*          dbl_f  = (float*)         (ws + 130678784);          // BL*96
    unsigned short* dtr_bf = (unsigned short*)(ws + 133824512);          // BL*R
    unsigned short* dt_bf  = (unsigned short*)(ws + 134873088);          // BL*E
    unsigned short* y_bf   = (unsigned short*)(ws + 168427520);          // BL*E

    // 1) residual add + LayerNorm
    resln_kernel<<<BL, 256, 0, stream>>>(x, residual, ln_w, ln_b, res_f, xn_bf);

    // 2) weight casts to bf16
    cast_f32_bf16<<<(2*Ee*Dm + 255) / 256, 256, 0, stream>>>(in_proj,  w_in,  2*Ee*Dm);
    cast_f32_bf16<<<((Rr+2*Nn)*Ee + 255) / 256, 256, 0, stream>>>(x_proj, w_xp, (Rr+2*Nn)*Ee);
    cast_f32_bf16<<<(Ee*Rr + 255) / 256, 256, 0, stream>>>(dt_proj, w_dt, Ee*Rr);
    cast_f32_bf16<<<(Dm*Ee + 255) / 256, 256, 0, stream>>>(out_proj, w_out, Dm*Ee);

    // 3) in_proj: xz = xn @ in_proj^T   (8192 x 4096 x 1024) -> bf16, JT=4 (block N=128)
    wmma_gemm_bf16<4, true><<<dim3((2*Ee)/128, BL/64), 256, 0, stream>>>(
        xn_bf, w_in, nullptr, xz_bf, BL, 2*Ee, Dm);

    // 4) depthwise causal conv + SiLU on xi
    conv_silu_kernel<<<(BL*Ee + 255) / 256, 256, 0, stream>>>(xz_bf, conv_w, conv_b, xc_bf);

    // 5) x_proj: dbl = xc @ x_proj^T    (8192 x 96 x 2048) -> f32, JT=3 (block N=96)
    wmma_gemm_bf16<3, false><<<dim3(1, BL/64), 256, 0, stream>>>(
        xc_bf, w_xp, dbl_f, nullptr, BL, Rr+2*Nn, Ee);

    // 6) slice dt-rank -> bf16
    extract_dt_kernel<<<(BL*Rr + 255) / 256, 256, 0, stream>>>(dbl_f, dtr_bf);

    // 7) dt_proj: dt = dtr @ dt_proj^T  (8192 x 2048 x 64) -> bf16, JT=4
    wmma_gemm_bf16<4, true><<<dim3(Ee/128, BL/64), 256, 0, stream>>>(
        dtr_bf, w_dt, nullptr, dt_bf, BL, Ee, Rr);

    // 8) selective scan + D-skip + SiLU(z) gate -> y bf16
    scan_kernel<<<dim3(Bb, Ee/256), 256, 0, stream>>>(
        dbl_f, dt_bf, xc_bf, xz_bf, dt_bias, A_log, D_skip, y_bf);

    // 9) out_proj: out = y @ out_proj^T (8192 x 1024 x 2048) -> f32 into d_out, JT=4
    wmma_gemm_bf16<4, false><<<dim3(Dm/128, BL/64), 256, 0, stream>>>(
        y_bf, w_out, out_f, nullptr, BL, Dm, Ee);
}
